// MarketTimeTransformer_46557445489092
// MI455X (gfx1250) — matrix operand order repaired and checked by reading, not compiled, not run on hardware
//
#include <hip/hip_runtime.h>
#include <hip/hip_bf16.h>
#include <math.h>

// ---------------- problem constants ----------------
#define PB 4
#define PS 2048
#define PDA 64
#define PDM 6
#define PD 256
#define PNH 8
#define PHD 32
#define PDFF 1024
#define PH 128
#define PBS (PB * PS)

#define BPITCH 40   // transposed B-tile LDS pitch in halves (80B: 16B-aligned, bank-spread)

typedef __attribute__((ext_vector_type(16))) _Float16 v16h;
typedef __attribute__((ext_vector_type(8)))  float    v8f;

// ---------------- small device helpers ----------------
__device__ __forceinline__ float geluf(float x) {
    return 0.5f * x * (1.0f + erff(x * 0.70710678118654752f));
}
__device__ __forceinline__ float softplusf(float x) {
    return (x > 20.0f) ? x : log1pf(expf(x));
}
__device__ __forceinline__ float sigmoidf(float x) {
    return 1.0f / (1.0f + expf(-x));
}
__device__ __forceinline__ float redmax16(float v) {
    #pragma unroll
    for (int m = 1; m < 16; m <<= 1) v = fmaxf(v, __shfl_xor(v, m, 32));
    return v;
}
__device__ __forceinline__ float redsum16(float v) {
    #pragma unroll
    for (int m = 1; m < 16; m <<= 1) v += __shfl_xor(v, m, 32);
    return v;
}
__device__ __forceinline__ float redsum32(float v) {
    #pragma unroll
    for (int m = 1; m < 32; m <<= 1) v += __shfl_xor(v, m, 32);
    return v;
}
__device__ __forceinline__ unsigned int pack2h(_Float16 a, _Float16 b) {
    unsigned short ua = __builtin_bit_cast(unsigned short, a);
    unsigned short ub = __builtin_bit_cast(unsigned short, b);
    return ((unsigned int)ub << 16) | (unsigned int)ua;
}

__device__ __forceinline__ v8f wmma_f16(v16h a, v16h b, v8f c) {
    // D = A(16x32 f16) x B(32x16 f16) + C(16x16 f32)
    return __builtin_amdgcn_wmma_f32_16x16x32_f16(
        false, a, false, b, (short)0, c, false, false);
}

// A fragment (16x32, f16), row-major LDS tile with pitch `ld`:
// lane l -> row M = l&15; K base = (l<16)?0:16; half j: K = Kbase + j
// Per lane this is two contiguous 16B runs -> 2x ds_load_b128.
__device__ __forceinline__ v16h ldsA(const _Float16* p, int ld, int row0, int lane) {
    int m  = lane & 15;
    int kb = (lane < 16) ? 0 : 16;
    const _Float16* q = p + (row0 + m) * ld + kb;
    v16h a;
    #pragma unroll
    for (int j = 0; j < 8; ++j)  a[j] = q[j];
    #pragma unroll
    for (int j = 8; j < 16; ++j) a[j] = q[j];
    return a;
}
// B fragment (32x16, f16) from a TRANSPOSED LDS tile Bs[col*BPITCH + k]:
// lane l -> col N = l&15; K base = (l<16)?0:16; half j: K = Kbase + j.
// Per lane: one contiguous 32B run -> 2x ds_load_b128.
__device__ __forceinline__ v16h ldsBT(const _Float16* p, int col0, int lane) {
    int n  = lane & 15;
    int kb = (lane < 16) ? 0 : 16;
    const _Float16* q = p + (col0 + n) * BPITCH + kb;
    v16h b;
    #pragma unroll
    for (int j = 0; j < 16; ++j) b[j] = q[j];
    return b;
}

// ---------------- kernel: market input linear (K=6, VALU) ----------------
__global__ void k_mk_ph(const float* __restrict__ mk, const float* __restrict__ w,
                        const float* __restrict__ bias, float* __restrict__ ph) {
    int idx = blockIdx.x * blockDim.x + threadIdx.x;
    if (idx >= PBS * PH) return;
    int h = idx % PH;
    int bs = idx / PH;
    float acc = bias[h];
    const float* m = mk + (size_t)bs * PDM;
    #pragma unroll
    for (int j = 0; j < PDM; ++j) acc += m[j] * w[j * PH + h];
    ph[idx] = acc;
}

// ---------------- kernel: dilated conv1d K=3 + GELU (+optional residual) ----
__global__ void k_conv(const float* __restrict__ in, const float* __restrict__ w,
                       const float* __restrict__ bias, const float* __restrict__ resid,
                       float* __restrict__ out, int dil) {
    int idx = blockIdx.x * blockDim.x + threadIdx.x;
    if (idx >= PBS * PH) return;
    int co = idx % PH;
    int s  = (idx / PH) % PS;
    int b  = idx / (PH * PS);
    float acc = bias[co];
    #pragma unroll
    for (int kk = 0; kk < 3; ++kk) {
        int ss = s + (kk - 1) * dil;
        if (ss < 0 || ss >= PS) continue;
        const float* ip = in + ((size_t)b * PS + ss) * PH;
        const float* wp = w + (size_t)kk * PH * PH + co;
        for (int ci = 0; ci < PH; ++ci) acc += ip[ci] * wp[ci * PH];
    }
    float g = geluf(acc);
    out[idx] = resid ? (resid[idx] + g) : g;
}

// ---------------- kernel: temporal mean summary ----------------
__global__ void k_summary(const float* __restrict__ ph, float* __restrict__ sm) {
    int b = blockIdx.x;
    int c = threadIdx.x;   // 128 threads
    float s = 0.f;
    for (int t = 0; t < PS; ++t) s += ph[((size_t)b * PS + t) * PH + c];
    sm[b * PH + c] = s * (1.0f / PS);
}

// ---------------- kernel: explicit-feature scan (serial per batch) ----------
__global__ void k_expterm(const float* __restrict__ mk, const float* __restrict__ alpha,
                          float* __restrict__ et) {
    int b = threadIdx.x;
    if (b >= PB) return;
    float a0 = alpha[0], a1 = alpha[1], a2 = alpha[2], a3 = alpha[3], a4 = alpha[4], a5 = alpha[5];
    float cum = 0.f, cr2 = 0.f, cmax = -1e30f, cturn = 0.f, prev = 0.f;
    for (int s = 0; s < PS; ++s) {
        const float* m = mk + ((size_t)b * PS + s) * PDM;
        float r = m[0], pos = m[2], inten = m[3];
        cum += r;
        cr2 += r * r;
        cmax = (s == 0) ? cum : fmaxf(cmax, cum);
        float dd = cum - cmax;
        float rv = sqrtf(cr2 / (float)(s + 1));
        float sg = (r > 0.f) ? 1.f : ((r < 0.f) ? -1.f : 0.f);
        float turn = (s == 0) ? 0.f : ((sg * prev < 0.f) ? 1.f : 0.f);
        cturn += turn;
        float tr = cturn / (float)(s + 1);
        et[b * PS + s] = a0 * cum + a1 * dd + a2 * rv + a3 * tr + a4 * pos + a5 * inten;
        prev = sg;
    }
}

// ---------------- kernel: concat(ph, broadcast summary) ----------------
__global__ void k_cat(const float* __restrict__ ph, const float* __restrict__ sm,
                      float* __restrict__ cat) {
    int idx = blockIdx.x * blockDim.x + threadIdx.x;
    if (idx >= PBS * 2 * PH) return;
    int c  = idx % (2 * PH);
    int s  = (idx / (2 * PH)) % PS;
    int b  = idx / (2 * PH * PS);
    cat[idx] = (c < PH) ? ph[((size_t)b * PS + s) * PH + c] : sm[b * PH + (c - PH)];
}

// ---------------- kernel: sr2 + softplus step ----------------
__global__ void k_step(const float* __restrict__ hres, const float* __restrict__ w2,
                       const float* __restrict__ b2, const float* __restrict__ et,
                       const float* __restrict__ escale, const float* __restrict__ lscale,
                       const float* __restrict__ base, float* __restrict__ step) {
    int idx = blockIdx.x * blockDim.x + threadIdx.x;
    if (idx >= PBS) return;
    const float* h = hres + (size_t)idx * PH;
    float d = b2[0];
    for (int c = 0; c < PH; ++c) d += h[c] * w2[c];
    float lr = tanhf(d);
    step[idx] = softplusf(base[0] + escale[0] * et[idx] + lscale[0] * lr);
}

// ---------------- kernel: tau cumsum (serial per batch) ----------------
__global__ void k_tau(const float* __restrict__ step, float* __restrict__ tau) {
    int b = threadIdx.x;
    if (b >= PB) return;
    float first = step[b * PS];
    float cum = 0.f;
    for (int s = 0; s < PS; ++s) {
        cum += step[b * PS + s];
        tau[b * PS + s] = cum - first;
    }
}

// ---------------- kernel: add economic-time positional encoding -------------
__global__ void k_pe(float* __restrict__ x, const float* __restrict__ tau,
                     const float* __restrict__ logit) {
    int idx = blockIdx.x * blockDim.x + threadIdx.x;
    if (idx >= PBS * PD) return;
    int d  = idx % PD;
    int bs = idx / PD;
    int j  = d >> 1;
    float div = expf(-(float)(2 * j) * (9.210340371976184f / (float)PD)); // ln(1e4)
    float ang = tau[bs] * div;
    float pe = (d & 1) ? cosf(ang) : sinf(ang);
    x[idx] += sigmoidf(logit[0]) * pe;
}

// ---------------- kernel: tau embedding sincos ----------------
__global__ void k_te(float* __restrict__ te, const float* __restrict__ tau) {
    int idx = blockIdx.x * blockDim.x + threadIdx.x;
    if (idx >= PBS * PD) return;
    int d  = idx % PD;
    int bs = idx / PD;
    int j  = d >> 1;
    float freq = 0.25f + (float)j * (1.75f / 127.0f);
    float ang = tau[bs] * freq;
    te[idx] = (d & 1) ? cosf(ang) : sinf(ang);
}

// ---------------- kernel: q/k gating combine ----------------
__global__ void k_gate(float* __restrict__ qb, const float* __restrict__ g,
                       const float* __restrict__ add) {
    int idx = blockIdx.x * blockDim.x + threadIdx.x;
    if (idx >= PBS * PD) return;
    qb[idx] = qb[idx] * (1.0f + g[idx]) + add[idx];
}

// ---------------- kernel: generic WMMA GEMM, C = act(A[MxK]*W[KxN]+bias) ----
// block = 128 threads (4 waves), block tile 64x64, K step 32.
// A tile row-major (pitch 32); B tile TRANSPOSED in LDS (Bs[col*BPITCH + k])
// so both fragment loads are contiguous -> ds_load_b128.
__global__ void k_gemm(const float* __restrict__ A, const float* __restrict__ W,
                       const float* __restrict__ bias, float* __restrict__ C,
                       int M, int K, int N, int act) {
    __shared__ _Float16 As[64 * 32];
    __shared__ _Float16 Bs[64 * BPITCH];
    int tid  = threadIdx.x;
    int wave = tid >> 5;
    int lane = tid & 31;
    int wm   = (wave >> 1) * 32;
    int wn   = (wave & 1) * 32;
    int row0 = blockIdx.y * 64;
    int col0 = blockIdx.x * 64;

    v8f acc00 = {}, acc01 = {}, acc10 = {}, acc11 = {};

    for (int k0 = 0; k0 < K; k0 += 32) {
        // ---- A tile: 64 rows x 32 k. float2 global loads, packed b32 LDS stores.
        for (int i = tid; i < 64 * 16; i += 128) {
            int r  = i >> 4;
            int cp = (i & 15) * 2;
            int gr = row0 + r;
            float f0 = 0.f, f1 = 0.f;
            if (gr < M) {
                float2 f = *(const float2*)(&A[(size_t)gr * K + k0 + cp]);
                f0 = f.x; f1 = f.y;
                if (k0 + 32 < K)
                    __builtin_prefetch(&A[(size_t)gr * K + k0 + 32 + cp], 0, 1);
            }
            *(unsigned int*)(&As[r * 32 + cp]) = pack2h((_Float16)f0, (_Float16)f1);
        }
        // ---- B tile: 32 k x 64 cols, stored transposed. Coalesced global reads
        //      (two consecutive k-rows swept across columns), packed b32 stores.
        for (int i = tid; i < 64 * 16; i += 128) {
            int rp = (i >> 6) * 2;   // even k within tile
            int c  = i & 63;         // column
            _Float16 h0 = (_Float16)W[(size_t)(k0 + rp) * N + col0 + c];
            _Float16 h1 = (_Float16)W[(size_t)(k0 + rp + 1) * N + col0 + c];
            *(unsigned int*)(&Bs[c * BPITCH + rp]) = pack2h(h0, h1);
        }
        __syncthreads();

        v16h a0 = ldsA(As, 32, wm,      lane);
        v16h a1 = ldsA(As, 32, wm + 16, lane);
        v16h b0 = ldsBT(Bs, wn,      lane);
        v16h b1 = ldsBT(Bs, wn + 16, lane);
        acc00 = wmma_f16(a0, b0, acc00);
        acc01 = wmma_f16(a0, b1, acc01);
        acc10 = wmma_f16(a1, b0, acc10);
        acc11 = wmma_f16(a1, b1, acc11);
        __syncthreads();
    }

    // epilogue: C layout - reg r -> row r + (lane<16?0:8); col = lane&15
    int n_in = lane & 15;
    int g8   = (lane < 16) ? 0 : 8;
    #pragma unroll
    for (int fj = 0; fj < 2; ++fj) {
        int col = col0 + wn + fj * 16 + n_in;
        float bv = bias[col];
        #pragma unroll
        for (int fi = 0; fi < 2; ++fi) {
            v8f acc = (fi == 0) ? (fj == 0 ? acc00 : acc01) : (fj == 0 ? acc10 : acc11);
            #pragma unroll
            for (int r = 0; r < 8; ++r) {
                int row = row0 + wm + fi * 16 + r + g8;
                if (row < M) {
                    float v = acc[r] + bv;
                    if (act == 1)      v = geluf(v);
                    else if (act == 2) v = tanhf(v);
                    C[(size_t)row * N + col] = v;
                }
            }
        }
    }
}

// ---------------- kernel: flash attention with tau-distance bias -----------
// one wave per (b, h, 16-query tile), streaming 32-key chunks.
__global__ void k_attn(const float* __restrict__ q, const float* __restrict__ k,
                       const float* __restrict__ v, const float* __restrict__ tau,
                       const float* __restrict__ slog, float* __restrict__ o) {
    __shared__ _Float16 pls[16 * 32];        // P tile, A-layout friendly
    __shared__ _Float16 Vs[32 * BPITCH];     // V chunk transposed: [hd][key]
    int qt = blockIdx.x, h = blockIdx.y, b = blockIdx.z;
    int lane = threadIdx.x;
    int q0 = qt * 16;
    float slope = softplusf(slog[h]);
    const float scale = 0.17677669529663687f; // 1/sqrt(32)

    int m  = lane & 15;
    int g8 = (lane < 16) ? 0 : 8;
    int n  = lane & 15;
    int kbA = (lane < 16) ? 0 : 16;

    // Q A-fragment (scale folded in): 64 contiguous bytes per lane from global
    const float* qp = q + ((size_t)b * PS + q0 + m) * PD + h * PHD;
    v16h aq;
    #pragma unroll
    for (int j = 0; j < 16; ++j) {
        int kk = kbA + j;
        aq[j] = (_Float16)(qp[kk] * scale);
    }

    // per-row online softmax state (rows r+g8 for this lane's registers)
    float tq[8], rmax[8], rsum[8];
    #pragma unroll
    for (int r = 0; r < 8; ++r) {
        tq[r]   = tau[(size_t)b * PS + q0 + r + g8];
        rmax[r] = -1e30f;
        rsum[r] = 0.f;
    }
    v8f acc0 = {}, acc1 = {};

    for (int kbase = 0; kbase < PS; kbase += 32) {
        // ---- stage V chunk transposed: lane = hd channel, sweep key pairs.
        //      Global side: 128B coalesced per key row; LDS side: packed b32.
        {
            const float* vbase = v + ((size_t)b * PS + kbase) * PD + h * PHD + lane;
            #pragma unroll
            for (int jp = 0; jp < 16; ++jp) {
                float f0 = vbase[(2 * jp) * PD];
                float f1 = vbase[(2 * jp + 1) * PD];
                *(unsigned int*)(&Vs[lane * BPITCH + 2 * jp]) = pack2h((_Float16)f0, (_Float16)f1);
            }
        }

        // ---- K^T B-fragments straight from global (contiguous 64B per lane)
        int cb = (lane < 16) ? 0 : 16;
        const float* kp0 = k + ((size_t)b * PS + kbase + n) * PD + h * PHD + cb;
        const float* kp1 = kp0 + 16 * PD;
        v16h bk0, bk1;
        #pragma unroll
        for (int j = 0; j < 16; ++j) {
            bk0[j] = (_Float16)kp0[j];
            bk1[j] = (_Float16)kp1[j];
        }
        v8f z = {};
        v8f s0 = wmma_f16(aq, bk0, z);
        v8f s1 = wmma_f16(aq, bk1, z);

        float tk0 = tau[(size_t)b * PS + kbase + n];
        float tk1 = tau[(size_t)b * PS + kbase + 16 + n];

        #pragma unroll
        for (int r = 0; r < 8; ++r) {
            float l0 = s0[r] - slope * fabsf(tq[r] - tk0);
            float l1 = s1[r] - slope * fabsf(tq[r] - tk1);
            float mloc = redmax16(fmaxf(l0, l1));
            float nm = fmaxf(rmax[r], mloc);
            float alpha = expf(rmax[r] - nm);
            float p0 = expf(l0 - nm);
            float p1 = expf(l1 - nm);
            rsum[r] = rsum[r] * alpha + redsum16(p0 + p1);
            acc0[r] *= alpha;
            acc1[r] *= alpha;
            rmax[r] = nm;
            pls[(r + g8) * 32 + n]      = (_Float16)p0;
            pls[(r + g8) * 32 + 16 + n] = (_Float16)p1;
        }
        __syncthreads();

        // P as A fragment (16 rows x 32 keys): one contiguous 32B run per lane
        const _Float16* pq = &pls[m * 32 + kbA];
        v16h ap;
        #pragma unroll
        for (int j = 0; j < 8; ++j)  ap[j] = pq[j];
        #pragma unroll
        for (int j = 8; j < 16; ++j) ap[j] = pq[j];

        // V B-fragments from transposed LDS: one contiguous 32B run per lane
        int kb2 = (lane < 16) ? 0 : 16;
        const _Float16* vq0 = &Vs[n * BPITCH + kb2];
        const _Float16* vq1 = &Vs[(16 + n) * BPITCH + kb2];
        v16h bv0, bv1;
        #pragma unroll
        for (int j = 0; j < 16; ++j) {
            bv0[j] = vq0[j];
            bv1[j] = vq1[j];
        }
        acc0 = wmma_f16(ap, bv0, acc0);
        acc1 = wmma_f16(ap, bv1, acc1);
        __syncthreads();
    }

    #pragma unroll
    for (int r = 0; r < 8; ++r) {
        float inv = 1.0f / rsum[r];
        size_t base = ((size_t)b * PS + q0 + r + g8) * PD + h * PHD;
        o[base + n]      = acc0[r] * inv;
        o[base + 16 + n] = acc1[r] * inv;
    }
}

// ---------------- kernel: residual add + layernorm (wave per row) ----------
__global__ void k_add_ln(const float* __restrict__ x, const float* __restrict__ y,
                         const float* __restrict__ g, const float* __restrict__ bta,
                         float* __restrict__ out) {
    int row = blockIdx.x;
    int l = threadIdx.x;
    const float* xr = x + (size_t)row * PD;
    const float* yr = y + (size_t)row * PD;
    float vals[8];
    float s = 0.f;
    #pragma unroll
    for (int i = 0; i < 8; ++i) {
        vals[i] = xr[l + 32 * i] + yr[l + 32 * i];
        s += vals[i];
    }
    float mean = redsum32(s) * (1.0f / PD);
    float vs = 0.f;
    #pragma unroll
    for (int i = 0; i < 8; ++i) {
        float d = vals[i] - mean;
        vs += d * d;
    }
    float rstd = rsqrtf(redsum32(vs) * (1.0f / PD) + 1e-5f);
    #pragma unroll
    for (int i = 0; i < 8; ++i) {
        int c = l + 32 * i;
        out[(size_t)row * PD + c] = (vals[i] - mean) * rstd * g[c] + bta[c];
    }
}

// ---------------- host launcher ----------------
extern "C" void kernel_launch(void* const* d_in, const int* in_sizes, int n_in,
                              void* d_out, int out_size, void* d_ws, size_t ws_size,
                              hipStream_t stream) {
    (void)in_sizes; (void)n_in; (void)out_size; (void)ws_size;
    const float* asset   = (const float*)d_in[0];
    const float* market  = (const float*)d_in[1];
    const float* mkw     = (const float*)d_in[2];
    const float* mkb     = (const float*)d_in[3];
    const float* c1w     = (const float*)d_in[4];
    const float* c1b     = (const float*)d_in[5];
    const float* c2w     = (const float*)d_in[6];
    const float* c2b     = (const float*)d_in[7];
    const float* sr1w    = (const float*)d_in[8];
    const float* sr1b    = (const float*)d_in[9];
    const float* sr2w    = (const float*)d_in[10];
    const float* sr2b    = (const float*)d_in[11];
    const float* alpha   = (const float*)d_in[12];
    const float* escale  = (const float*)d_in[13];
    const float* lscale  = (const float*)d_in[14];
    const float* baselog = (const float*)d_in[15];
    const float* pelogit = (const float*)d_in[16];
    const float* aw      = (const float*)d_in[17];
    const float* ab      = (const float*)d_in[18];
    const float* qw  = (const float*)d_in[19], *qbi = (const float*)d_in[20];
    const float* kw  = (const float*)d_in[21], *kbi = (const float*)d_in[22];
    const float* vw  = (const float*)d_in[23], *vbi = (const float*)d_in[24];
    const float* ow  = (const float*)d_in[25], *obi = (const float*)d_in[26];
    const float* qgw = (const float*)d_in[27], *qgb = (const float*)d_in[28];
    const float* qbw = (const float*)d_in[29], *qbb = (const float*)d_in[30];
    const float* kgw = (const float*)d_in[31], *kgb = (const float*)d_in[32];
    const float* kbw = (const float*)d_in[33], *kbb = (const float*)d_in[34];
    const float* slog = (const float*)d_in[35];
    const float* f1w = (const float*)d_in[36], *f1b = (const float*)d_in[37];
    const float* f2w = (const float*)d_in[38], *f2b = (const float*)d_in[39];
    const float* n1g = (const float*)d_in[40], *n1b = (const float*)d_in[41];
    const float* n2g = (const float*)d_in[42], *n2b = (const float*)d_in[43];

    float* ws = (float*)d_ws;
    size_t off = 0;
    auto alloc = [&](size_t n) { float* p = ws + off; off += n; return p; };
    const size_t BSH = (size_t)PBS * PH;
    const size_t BSD = (size_t)PBS * PD;
    float* ph   = alloc(BSH);
    float* cb1  = alloc(BSH);
    float* phf  = alloc(BSH);
    float* smry = alloc(PB * PH);
    float* et   = alloc(PBS);
    float* cat  = alloc((size_t)PBS * 2 * PH);
    float* hres = alloc(BSH);
    float* stp  = alloc(PBS);
    float* tau  = alloc(PBS);
    float* x    = alloc(BSD);
    float* te   = alloc(BSD);
    float* qv   = alloc(BSD);
    float* kv   = alloc(BSD);
    float* vv   = alloc(BSD);
    float* tA   = alloc(BSD);
    float* tB   = alloc(BSD);
    float* obuf = alloc(BSD);
    float* x1   = alloc(BSD);
    float* ffh  = alloc((size_t)PBS * PDFF);

    auto gemm = [&](const float* A, const float* W, const float* bias, float* C,
                    int M, int K, int N, int act) {
        k_gemm<<<dim3(N / 64, (M + 63) / 64), 128, 0, stream>>>(A, W, bias, C, M, K, N, act);
    };

    int nBSH = (int)((BSH + 255) / 256);
    int nBSD = (int)((BSD + 255) / 256);

    // market trajectory encoder
    k_mk_ph<<<nBSH, 256, 0, stream>>>(market, mkw, mkb, ph);
    k_conv<<<nBSH, 256, 0, stream>>>(ph, c1w, c1b, nullptr, cb1, 1);
    k_conv<<<nBSH, 256, 0, stream>>>(cb1, c2w, c2b, ph, phf, 2);
    k_summary<<<PB, PH, 0, stream>>>(phf, smry);
    k_expterm<<<1, PB, 0, stream>>>(market, alpha, et);
    k_cat<<<(int)(((size_t)PBS * 2 * PH + 255) / 256), 256, 0, stream>>>(phf, smry, cat);
    gemm(cat, sr1w, sr1b, hres, PBS, 2 * PH, PH, 1);
    k_step<<<(PBS + 255) / 256, 256, 0, stream>>>(hres, sr2w, sr2b, et, escale, lscale, baselog, stp);
    k_tau<<<1, PB, 0, stream>>>(stp, tau);

    // asset embedding + positional encoding + tau embedding
    gemm(asset, aw, ab, x, PBS, PDA, PD, 0);
    k_pe<<<nBSD, 256, 0, stream>>>(x, tau, pelogit);
    k_te<<<nBSD, 256, 0, stream>>>(te, tau);

    // projections + tau-conditioned gating
    gemm(x, qw, qbi, qv, PBS, PD, PD, 0);
    gemm(x, kw, kbi, kv, PBS, PD, PD, 0);
    gemm(x, vw, vbi, vv, PBS, PD, PD, 0);
    gemm(te, qgw, qgb, tA, PBS, PD, PD, 2);
    gemm(te, qbw, qbb, tB, PBS, PD, PD, 0);
    k_gate<<<nBSD, 256, 0, stream>>>(qv, tA, tB);
    gemm(te, kgw, kgb, tA, PBS, PD, PD, 2);
    gemm(te, kbw, kbb, tB, PBS, PD, PD, 0);
    k_gate<<<nBSD, 256, 0, stream>>>(kv, tA, tB);

    // flash attention with tau-distance bias
    k_attn<<<dim3(PS / 16, PNH, PB), 32, 0, stream>>>(qv, kv, vv, tau, slog, obuf);

    // output proj + add&norm + FFN + add&norm
    gemm(obuf, ow, obi, tA, PBS, PD, PD, 0);
    k_add_ln<<<PBS, 32, 0, stream>>>(x, tA, n1g, n1b, x1);
    gemm(x1, f1w, f1b, ffh, PBS, PD, PDFF, 1);
    gemm(ffh, f2w, f2b, tB, PBS, PDFF, PD, 0);
    k_add_ln<<<PBS, 32, 0, stream>>>(x1, tB, n2g, n2b, (float*)d_out);
}
